// VisionMamba_27230092656986
// MI455X (gfx1250) — matrix operand order repaired and checked by reading, not compiled, not run on hardware
//
#include <hip/hip_runtime.h>
#include <hip/hip_bf16.h>

// ---------------------------------------------------------------------------
// Vision Mamba forward for MI455X (gfx1250, wave32, WMMA).
// - GEMMs: v_wmma_f32_16x16x32_bf16, bf16 operands pre-converted once,
//   1x4 output-tile strips per wave (A-fragment reuse).
// - Scan: whole dbc strip staged to LDS via async global->LDS (ASYNCcnt),
//   then 197-step recurrence runs barrier-free on v_exp_f32 + ds_load_b128.
// ---------------------------------------------------------------------------

typedef __attribute__((ext_vector_type(16))) __bf16 v16bf;
typedef __attribute__((ext_vector_type(8)))  __bf16 v8bf;
typedef __attribute__((ext_vector_type(8)))  float  v8f;
typedef __attribute__((ext_vector_type(4)))  int    v4i;

#define BSZ     32
#define DEPTH_  24
#define DD      192
#define EE      384
#define NN      16
#define RR_     12
#define LPATCH  196
#define LSEQ    197
#define CLSP    98
#define NCLS    1000
#define ROWS    (BSZ * LSEQ)    // 6304 (divisible by 16)
#define PROWS   (BSZ * LPATCH)  // 6272 (divisible by 16)

#if __has_builtin(__builtin_amdgcn_global_load_async_to_lds_b128) && \
    __has_builtin(__builtin_amdgcn_s_wait_asynccnt)
#define HAVE_ASYNC_LDS 1
#endif

__device__ __forceinline__ __bf16 tobf(float f) {
  unsigned u = __builtin_bit_cast(unsigned, f);
  unsigned r = u + 0x7FFFu + ((u >> 16) & 1u);
  unsigned short hbits = (unsigned short)(r >> 16);
  return __builtin_bit_cast(__bf16, hbits);
}
__device__ __forceinline__ float frombf(__bf16 h) {
  unsigned u = ((unsigned)__builtin_bit_cast(unsigned short, h)) << 16;
  return __builtin_bit_cast(float, u);
}
__device__ __forceinline__ float sigm(float x)  { return 1.f / (1.f + __expf(-x)); }
__device__ __forceinline__ float softp(float x) { return x > 20.f ? x : __logf(1.f + __expf(x)); }

// --------------------------- fp32 -> bf16 bulk convert ---------------------
__global__ void cvt_bf16_kernel(const float* __restrict__ src,
                                __bf16* __restrict__ dst, int n) {
  int i = blockIdx.x * blockDim.x + threadIdx.x;
  if (i < n) dst[i] = tobf(src[i]);
}

// --------------------------- generic WMMA GEMM (bf16 in, f32 out) ----------
// C[M x ldc] = A[M x K] * W[N x K]^T ; one 16 x (16*NTW) tile strip per wave.
template <int NTW>
__global__ void gemm_bf16_wmma(const __bf16* __restrict__ A,
                               const __bf16* __restrict__ W,
                               float* __restrict__ C,
                               int M, int K, int NtGroups, int Nvalid, int ldc) {
  int wave = (blockIdx.x * blockDim.x + threadIdx.x) >> 5;
  int lane = threadIdx.x & 31;
  int Mtiles = M >> 4;
  if (wave >= Mtiles * NtGroups) return;
  int mt = wave / NtGroups, ng = wave % NtGroups;
  int m0 = mt << 4, n0 = (ng * NTW) << 4;
  int hv = lane >> 4, l15 = lane & 15;

  v8f acc[NTW];
#pragma unroll
  for (int j = 0; j < NTW; ++j) acc[j] = {};

  const __bf16* arow = A + (size_t)(m0 + l15) * K + hv * 16;
  for (int k0 = 0; k0 < K; k0 += 32) {
    if (k0 + 64 < K) __builtin_prefetch(arow + k0 + 64, 0, 3);
    v8bf alo = *(const v8bf*)(arow + k0);
    v8bf ahi = *(const v8bf*)(arow + k0 + 8);
    v16bf a = __builtin_shufflevector(alo, ahi, 0, 1, 2, 3, 4, 5, 6, 7,
                                      8, 9, 10, 11, 12, 13, 14, 15);
#pragma unroll
    for (int j = 0; j < NTW; ++j) {
      int n = n0 + j * 16 + l15;
      v16bf b = {};
      if (n < Nvalid) b = *(const v16bf*)(W + (size_t)n * K + k0 + hv * 16);
      acc[j] = __builtin_amdgcn_wmma_f32_16x16x32_bf16(false, a, false, b,
                                                       (short)0, acc[j], false, false);
    }
  }
  int rbase = m0 + (hv << 3);
#pragma unroll
  for (int j = 0; j < NTW; ++j) {
    int n = n0 + j * 16 + l15;
    if (n < Nvalid) {
#pragma unroll
      for (int r = 0; r < 8; ++r) C[(size_t)(rbase + r) * ldc + n] = acc[j][r];
    }
  }
}

// --------------------------- im2col of input patches to bf16 ---------------
__global__ void im2col_kernel(const float* __restrict__ x, __bf16* __restrict__ a) {
  int idx = blockIdx.x * blockDim.x + threadIdx.x;
  if (idx >= PROWS * 768) return;
  int row = idx / 768, kk = idx % 768;
  int bb = row / LPATCH, l = row % LPATCH;
  int ly = l / 14, lx = l % 14;
  int c = kk >> 8, rem = kk & 255, py = rem >> 4, px = rem & 15;
  a[idx] = tobf(x[(((size_t)bb * 3 + c) * 224 + (ly * 16 + py)) * 224 + (lx * 16 + px)]);
}

// --------------------------- patch embedding GEMM (bf16 WMMA) --------------
__global__ void patch_gemm_kernel(const __bf16* __restrict__ A,   // [PROWS x 768]
                                  const __bf16* __restrict__ W,   // [192 x 768]
                                  const float* __restrict__ pb,
                                  const float* __restrict__ pos,
                                  float* __restrict__ hidden) {
  const int NTW = 4, NtGroups = 3;                 // 12 N-tiles
  int wave = (blockIdx.x * blockDim.x + threadIdx.x) >> 5;
  int lane = threadIdx.x & 31;
  if (wave >= (PROWS / 16) * NtGroups) return;
  int mt = wave / NtGroups, ng = wave % NtGroups;
  int m0 = mt << 4, n0 = (ng * NTW) << 4;
  int hv = lane >> 4, l15 = lane & 15;

  v8f acc[NTW];
#pragma unroll
  for (int j = 0; j < NTW; ++j) acc[j] = {};

  const __bf16* arow = A + (size_t)(m0 + l15) * 768 + hv * 16;
  for (int k0 = 0; k0 < 768; k0 += 32) {
    if (k0 + 64 < 768) __builtin_prefetch(arow + k0 + 64, 0, 3);
    v8bf alo = *(const v8bf*)(arow + k0);
    v8bf ahi = *(const v8bf*)(arow + k0 + 8);
    v16bf a = __builtin_shufflevector(alo, ahi, 0, 1, 2, 3, 4, 5, 6, 7,
                                      8, 9, 10, 11, 12, 13, 14, 15);
#pragma unroll
    for (int j = 0; j < NTW; ++j) {
      int n = n0 + j * 16 + l15;
      v16bf b = *(const v16bf*)(W + (size_t)n * 768 + k0 + hv * 16);
      acc[j] = __builtin_amdgcn_wmma_f32_16x16x32_bf16(false, a, false, b,
                                                       (short)0, acc[j], false, false);
    }
  }
  int rbase = m0 + (hv << 3);
#pragma unroll
  for (int j = 0; j < NTW; ++j) {
    int n = n0 + j * 16 + l15;
#pragma unroll
    for (int r = 0; r < 8; ++r) {
      int row = rbase + r;
      int b2 = row / LPATCH, l2 = row % LPATCH;
      int p = (l2 < CLSP) ? l2 : l2 + 1;           // TOKEN_IDX: cls inserted at 98
      hidden[((size_t)b2 * LSEQ + p) * DD + n] =
          acc[j][r] + pb[n] + pos[(size_t)l2 * DD + n];
    }
  }
}

__global__ void cls_kernel(const float* __restrict__ cls,
                           const float* __restrict__ posc,
                           float* __restrict__ hidden) {
  int b = blockIdx.x, d = threadIdx.x;
  hidden[((size_t)b * LSEQ + CLSP) * DD + d] = cls[d] + posc[d];
}

// --------------------------- residual add + LayerNorm (wave/token) ---------
__global__ void resln_kernel(float* __restrict__ residual,
                             const float* __restrict__ hidden,
                             __bf16* __restrict__ hn,
                             const float* __restrict__ w,
                             const float* __restrict__ b) {
  int wave = (blockIdx.x * blockDim.x + threadIdx.x) >> 5;
  int lane = threadIdx.x & 31;
  if (wave >= ROWS) return;
  size_t base = (size_t)wave * DD;
  float v[6];
  float s = 0.f;
#pragma unroll
  for (int i = 0; i < 6; ++i) {
    int j = lane * 6 + i;
    float t = residual[base + j] + hidden[base + j];
    residual[base + j] = t;
    v[i] = t; s += t;
  }
#pragma unroll
  for (int off = 16; off > 0; off >>= 1) s += __shfl_xor(s, off, 32);
  float mean = s * (1.f / DD);
  float vs = 0.f;
#pragma unroll
  for (int i = 0; i < 6; ++i) { float d = v[i] - mean; vs += d * d; }
#pragma unroll
  for (int off = 16; off > 0; off >>= 1) vs += __shfl_xor(vs, off, 32);
  float inv = rsqrtf(vs * (1.f / DD) + 1e-5f);
#pragma unroll
  for (int i = 0; i < 6; ++i) {
    int j = lane * 6 + i;
    hn[base + j] = tobf((v[i] - mean) * inv * w[j] + b[j]);
  }
}

// --------------------------- causal conv (DC=4) + SiLU, per direction ------
__global__ void conv_silu_kernel(const float* __restrict__ xz,
                                 const float* __restrict__ cw,
                                 const float* __restrict__ cb,
                                 __bf16* __restrict__ xc, int dir) {
  int idx = blockIdx.x * blockDim.x + threadIdx.x;
  if (idx >= ROWS * EE) return;
  int e = idx % EE;
  int t = (idx / EE) % LSEQ;
  int bb = idx / (EE * LSEQ);
  float acc = cb[e];
#pragma unroll
  for (int k = 0; k < 4; ++k) {
    int ts = t + k - 3;
    if (ts >= 0) {
      int tm = dir ? (LSEQ - 1 - ts) : ts;          // flipped sequence for bwd
      acc += xz[((size_t)bb * LSEQ + tm) * (2 * EE) + e] * cw[e * 4 + k];
    }
  }
  xc[idx] = tobf(acc * sigm(acc));
}

// --------------------------- selective-scan: state in VGPRs ----------------
// grid (BSZ, 2); block EE threads (12 waves); lane = channel e.
// Whole (b,dir) dbc strip (197 x 48 floats = 37.8 KB) is staged to LDS up
// front via async global->LDS; recurrence then runs without barriers.
__global__ void scan_kernel(const __bf16* __restrict__ xcf,
                            const __bf16* __restrict__ xcb,
                            const float* __restrict__ dbcf,
                            const float* __restrict__ dbcb,
                            const float* __restrict__ dtw,
                            const float* __restrict__ dtb,
                            const float* __restrict__ alog,
                            const float* __restrict__ Dp,
                            float* __restrict__ yfo,
                            float* __restrict__ ybo) {
  int bb = blockIdx.x, dir = blockIdx.y;
  int e = threadIdx.x;
  const __bf16* xc  = dir ? xcb  : xcf;
  const float*  dbc = dir ? dbcb : dbcf;
  float*        y   = dir ? ybo  : yfo;

  __shared__ float sdbc[LSEQ * 48];
  {
    const float* src = dbc + (size_t)bb * LSEQ * 48;
    const int nvec = (LSEQ * 48) / 4;               // 2364 float4s
#ifdef HAVE_ASYNC_LDS
    for (int i = threadIdx.x; i < nvec; i += EE) {
      __builtin_amdgcn_global_load_async_to_lds_b128(
          (v4i*)(src + i * 4), (v4i*)(sdbc + i * 4), 0, 0);
    }
    __builtin_amdgcn_s_wait_asynccnt(0);
#else
    for (int i = threadIdx.x; i < nvec; i += EE)
      *(float4*)(sdbc + i * 4) = *(const float4*)(src + i * 4);
#endif
    __syncthreads();
  }

  float Ae[NN], hstate[NN], wdt[RR_];
#pragma unroll
  for (int n = 0; n < NN; ++n) { Ae[n] = -__expf(alog[(size_t)e * NN + n]); hstate[n] = 0.f; }
#pragma unroll
  for (int r = 0; r < RR_; ++r) wdt[r] = dtw[(size_t)e * RR_ + r];
  float bias = dtb[e], dpe = Dp[e];

  for (int t = 0; t < LSEQ; ++t) {
    size_t row = (size_t)bb * LSEQ + t;
    const float4* r4 = (const float4*)(sdbc + t * 48);  // 16B-aligned (192B rows)
    float dtv[12], Bs[NN], Cs[NN];
    *(float4*)&dtv[0] = r4[0]; *(float4*)&dtv[4] = r4[1]; *(float4*)&dtv[8]  = r4[2];
    *(float4*)&Bs[0]  = r4[3]; *(float4*)&Bs[4]  = r4[4]; *(float4*)&Bs[8]   = r4[5];
    *(float4*)&Bs[12] = r4[6];
    *(float4*)&Cs[0]  = r4[7]; *(float4*)&Cs[4]  = r4[8]; *(float4*)&Cs[8]   = r4[9];
    *(float4*)&Cs[12] = r4[10];

    float dtr = bias;
#pragma unroll
    for (int r = 0; r < RR_; ++r) dtr += dtv[r] * wdt[r];
    float dt = softp(dtr);
    float xt = frombf(xc[row * EE + e]);
    float dtx = dt * xt;
    float yacc = 0.f;
#pragma unroll
    for (int n = 0; n < NN; ++n) {
      float dA = __expf(dt * Ae[n]);
      hstate[n] = dA * hstate[n] + dtx * Bs[n];
      yacc += hstate[n] * Cs[n];
    }
    y[row * EE + e] = yacc + dpe * xt;               // skip-connection x*D
  }
}

// --------------------------- gate: (yf + flip(yb)) * silu(z) -> bf16 -------
__global__ void gate_kernel(const float* __restrict__ yf,
                            const float* __restrict__ yb,
                            const float* __restrict__ xz,
                            __bf16* __restrict__ g) {
  int idx = blockIdx.x * blockDim.x + threadIdx.x;
  if (idx >= ROWS * EE) return;
  int e = idx % EE;
  int t = (idx / EE) % LSEQ;
  int bb = idx / (EE * LSEQ);
  size_t row  = (size_t)bb * LSEQ + t;
  size_t rowf = (size_t)bb * LSEQ + (LSEQ - 1 - t);
  float z = xz[row * (2 * EE) + EE + e];
  g[row * EE + e] = tobf((yf[row * EE + e] + yb[rowf * EE + e]) * (z * sigm(z)));
}

// --------------------------- classification head (WMMA, gathered rows) -----
__global__ void head_kernel(const __bf16* __restrict__ hnb,
                            const __bf16* __restrict__ hw,
                            const float* __restrict__ hb,
                            float* __restrict__ out) {
  int wave = (blockIdx.x * blockDim.x + threadIdx.x) >> 5;
  int lane = threadIdx.x & 31;
  const int Ntiles = 63;                 // covers 1008 >= 1000
  if (wave >= 2 * Ntiles) return;
  int mt = wave / Ntiles, nt = wave % Ntiles;
  int m0 = mt << 4, n0 = nt << 4;
  int hv = lane >> 4, l15 = lane & 15;

  v8f acc = {};
  int ncol = n0 + l15;
  int brow = m0 + l15;                   // batch index 0..31
  const __bf16* arow = hnb + ((size_t)brow * LSEQ + CLSP) * DD + hv * 16;
  for (int k0 = 0; k0 < DD; k0 += 32) {
    v8bf alo = *(const v8bf*)(arow + k0);
    v8bf ahi = *(const v8bf*)(arow + k0 + 8);
    v16bf a = __builtin_shufflevector(alo, ahi, 0, 1, 2, 3, 4, 5, 6, 7,
                                      8, 9, 10, 11, 12, 13, 14, 15);
    v16bf b = {};
    if (ncol < NCLS) b = *(const v16bf*)(hw + (size_t)ncol * DD + k0 + hv * 16);
    acc = __builtin_amdgcn_wmma_f32_16x16x32_bf16(false, a, false, b,
                                                  (short)0, acc, false, false);
  }
  if (ncol < NCLS) {
    int rbase = m0 + (hv << 3);
#pragma unroll
    for (int r = 0; r < 8; ++r)
      out[(size_t)(rbase + r) * NCLS + ncol] = acc[r] + hb[ncol];
  }
}

// ---------------------------------------------------------------------------
static void cvt(const float* s, __bf16* d, int n, hipStream_t st) {
  cvt_bf16_kernel<<<(n + 255) / 256, 256, 0, st>>>(s, d, n);
}
template <int NTW>
static void launch_gemm(const __bf16* A, const __bf16* W, float* C, int M, int K,
                        int NtGroups, int Nvalid, int ldc, hipStream_t s) {
  int waves = (M / 16) * NtGroups;
  gemm_bf16_wmma<NTW><<<(waves + 7) / 8, 256, 0, s>>>(A, W, C, M, K, NtGroups, Nvalid, ldc);
}

extern "C" void kernel_launch(void* const* d_in, const int* in_sizes, int n_in,
                              void* d_out, int out_size, void* d_ws, size_t ws_size,
                              hipStream_t stream) {
  (void)in_sizes; (void)n_in; (void)out_size; (void)ws_size;
  const float* x       = (const float*)d_in[0];
  const float* patch_w = (const float*)d_in[1];
  const float* patch_b = (const float*)d_in[2];
  const float* pos     = (const float*)d_in[3];
  const float* cls     = (const float*)d_in[4];
  const float* posc    = (const float*)d_in[5];
  const float* norm_w  = (const float*)d_in[6];
  const float* norm_b  = (const float*)d_in[7];
  const float* in_w    = (const float*)d_in[8];
  const float* conv_w  = (const float*)d_in[9];
  const float* conv_b  = (const float*)d_in[10];
  const float* xproj_w = (const float*)d_in[11];
  const float* dt_w    = (const float*)d_in[12];
  const float* dt_b    = (const float*)d_in[13];
  const float* A_log   = (const float*)d_in[14];
  const float* Dp      = (const float*)d_in[15];
  const float* out_w   = (const float*)d_in[16];
  const float* normf_w = (const float*)d_in[17];
  const float* normf_b = (const float*)d_in[18];
  const float* head_w  = (const float*)d_in[19];
  const float* head_b  = (const float*)d_in[20];
  float* out = (float*)d_out;

  // ---- fp32 scratch ----
  float* ws = (float*)d_ws;
  size_t o = 0;
  float* residual = ws + o; o += (size_t)ROWS * DD;
  float* hidden   = ws + o; o += (size_t)ROWS * DD;
  float* xz       = ws + o; o += (size_t)ROWS * 2 * EE;
  float* dbcf     = ws + o; o += (size_t)ROWS * 48;
  float* dbcb     = ws + o; o += (size_t)ROWS * 48;
  float* yf       = ws + o; o += (size_t)ROWS * EE;
  float* yb       = ws + o; o += (size_t)ROWS * EE;
  // ---- bf16 scratch ----
  __bf16* bws = (__bf16*)(ws + o);
  size_t ob = 0;
  __bf16* wbf_in    = bws + ob; ob += (size_t)DEPTH_ * 2 * EE * DD;   // 24x768x192
  __bf16* wbf_xp    = bws + ob; ob += (size_t)DEPTH_ * 44 * EE;       // 24x44x384
  __bf16* wbf_out   = bws + ob; ob += (size_t)DEPTH_ * DD * EE;       // 24x192x384
  __bf16* wbf_head  = bws + ob; ob += (size_t)NCLS * DD;              // 1000x192
  __bf16* wbf_patch = bws + ob; ob += (size_t)DD * 768;               // 192x768
  __bf16* im2col    = bws + ob; ob += (size_t)PROWS * 768;
  __bf16* hn_bf     = bws + ob; ob += (size_t)ROWS * DD;
  __bf16* xcf_bf    = bws + ob; ob += (size_t)ROWS * EE;
  __bf16* xcb_bf    = bws + ob; ob += (size_t)ROWS * EE;
  __bf16* g_bf      = bws + ob; ob += (size_t)ROWS * EE;

  // ---- one-time (per launch) weight conversion to bf16 ----
  cvt(in_w,    wbf_in,    DEPTH_ * 2 * EE * DD, stream);
  cvt(xproj_w, wbf_xp,    DEPTH_ * 44 * EE,     stream);
  cvt(out_w,   wbf_out,   DEPTH_ * DD * EE,     stream);
  cvt(head_w,  wbf_head,  NCLS * DD,            stream);
  cvt(patch_w, wbf_patch, DD * 768,             stream);

  (void)hipMemsetAsync(residual, 0, (size_t)ROWS * DD * sizeof(float), stream);

  // ---- patch embedding ----
  im2col_kernel<<<(PROWS * 768 + 255) / 256, 256, 0, stream>>>(x, im2col);
  {
    int waves = (PROWS / 16) * 3;
    patch_gemm_kernel<<<(waves + 7) / 8, 256, 0, stream>>>(im2col, wbf_patch,
                                                           patch_b, pos, hidden);
    cls_kernel<<<BSZ, DD, 0, stream>>>(cls, posc, hidden);
  }

  int ew_blocks = (ROWS * EE + 255) / 256;
  for (int i = 0; i < DEPTH_; ++i) {
    resln_kernel<<<ROWS / 8, 256, 0, stream>>>(residual, hidden, hn_bf,
                                               norm_w + i * DD, norm_b + i * DD);
    launch_gemm<4>(hn_bf, wbf_in + (size_t)i * 2 * EE * DD, xz,
                   ROWS, DD, 12, 2 * EE, 2 * EE, stream);
    conv_silu_kernel<<<ew_blocks, 256, 0, stream>>>(xz, conv_w + (size_t)i * EE * 4,
                                                    conv_b + i * EE, xcf_bf, 0);
    conv_silu_kernel<<<ew_blocks, 256, 0, stream>>>(xz, conv_w + (size_t)i * EE * 4,
                                                    conv_b + i * EE, xcb_bf, 1);
    launch_gemm<3>(xcf_bf, wbf_xp + (size_t)i * 44 * EE, dbcf,
                   ROWS, EE, 1, 44, 48, stream);
    launch_gemm<3>(xcb_bf, wbf_xp + (size_t)i * 44 * EE, dbcb,
                   ROWS, EE, 1, 44, 48, stream);
    scan_kernel<<<dim3(BSZ, 2), EE, 0, stream>>>(xcf_bf, xcb_bf, dbcf, dbcb,
                                                 dt_w + (size_t)i * EE * RR_,
                                                 dt_b + i * EE,
                                                 A_log + (size_t)i * EE * NN,
                                                 Dp + i * EE, yf, yb);
    gate_kernel<<<ew_blocks, 256, 0, stream>>>(yf, yb, xz, g_bf);
    launch_gemm<4>(g_bf, wbf_out + (size_t)i * DD * EE, hidden,
                   ROWS, EE, 3, DD, DD, stream);
  }

  resln_kernel<<<ROWS / 8, 256, 0, stream>>>(residual, hidden, hn_bf, normf_w, normf_b);
  {
    int waves = 2 * 63;
    head_kernel<<<(waves * 32 + 255) / 256, 256, 0, stream>>>(hn_bf, wbf_head, head_b, out);
  }
}
